// TensoRFRenderer_188978561547
// MI455X (gfx1250) — compile-verified, matrix operand order checked by reading, not run on hardware
//
#include <hip/hip_runtime.h>
#include <hip/hip_bf16.h>

typedef __attribute__((ext_vector_type(16))) _Float16 v16h;
typedef __attribute__((ext_vector_type(8)))  float    v8f;

#define NRAYS      8192
#define NSAMP      173            // int(sqrt(3)*3/STEP)+1
#define CHUNK      32
#define NCHUNK     6              // 6*32 = 192 >= 173
#define WAVES      4              // waves (rays) per block
#define HPAD       66             // padded LDS row (64 + 2) in halves

static constexpr float STEPF   = (float)(3.0 / (300 - 1 + 0.001) * 3.0);
static constexpr float DSCALE  = 25.0f;
static constexpr float WTHRES  = 1.0e-4f;

// out layout: rgb[8192*3] | sem[8192*20] | inst[8192*16] | depth[8192] | dist_reg[1]
#define SEM_BASE   (NRAYS*3)
#define INST_BASE  (SEM_BASE + NRAYS*20)
#define DEPTH_BASE (INST_BASE + NRAYS*16)
#define DREG_IDX   (DEPTH_BASE + NRAYS)

__device__ __forceinline__ float wred(float x) {
  #pragma unroll
  for (int o = 16; o >= 1; o >>= 1) x += __shfl_xor(x, o, 32);
  return x;
}

// ---------------------------------------------------------------------------
// Prep: fold Wapp into Wa1 (-> Wfold 6x64), pad Wd1 to K=32 (f16),
//       combine Wsf@Wsm (-> Wcomb 3x20), zero the dist_reg accumulator.
// ---------------------------------------------------------------------------
__global__ void prep_kernel(const float* __restrict__ Wd1,  const float* __restrict__ Wapp,
                            const float* __restrict__ Wa1,  const float* __restrict__ Wsf,
                            const float* __restrict__ Wsm,
                            _Float16* __restrict__ Wd1h, _Float16* __restrict__ Wfoldh,
                            float* __restrict__ Wcomb, float* __restrict__ dreg)
{
  const int tid = threadIdx.x;
  for (int idx = tid; idx < 32*64; idx += blockDim.x) {
    const int k = idx >> 6, c = idx & 63;
    Wd1h[idx] = (k < 3) ? (_Float16)Wd1[k*64 + c] : (_Float16)0.0f;
    float f = 0.0f;
    if (k < 3) {
      for (int j = 0; j < 27; ++j) f += Wapp[k*27 + j] * Wa1[j*64 + c];
    } else if (k < 6) {
      f = Wa1[(24 + k)*64 + c];          // rows 27..29 of Wa1 (viewdir part)
    }
    Wfoldh[idx] = (_Float16)f;
  }
  for (int idx = tid; idx < 3*20; idx += blockDim.x) {
    const int r = idx / 20, c = idx % 20;
    float f = 0.0f;
    for (int j = 0; j < 32; ++j) f += Wsf[r*32 + j] * Wsm[j*20 + c];
    Wcomb[idx] = f;
  }
  if (tid == 0) *dreg = 0.0f;
}

// ---------------------------------------------------------------------------
// Fused renderer: one wave per ray, WMMA for both 64-wide layers,
// shuffle-scan alpha compositing, per-wave output reduction.
// ---------------------------------------------------------------------------
__global__ __launch_bounds__(WAVES*32)
void render_kernel(const float* __restrict__ rays,
                   const float* __restrict__ bd1, const float* __restrict__ Wd2,
                   const float* __restrict__ bd2, const float* __restrict__ ba1,
                   const float* __restrict__ Wa2, const float* __restrict__ ba2,
                   const float* __restrict__ bsm, const float* __restrict__ Wi,
                   const _Float16* __restrict__ Wd1h, const _Float16* __restrict__ Wfoldh,
                   const float* __restrict__ Wcomb, const int* __restrict__ white_bg,
                   float* __restrict__ out)
{
  __shared__ _Float16 Hden[WAVES][CHUNK][HPAD];
  __shared__ _Float16 Happ[WAVES][CHUNK][HPAD];

  const int lane = threadIdx.x & 31;
  const int wv   = threadIdx.x >> 5;
  const int ray  = blockIdx.x * WAVES + wv;
  const int lh   = lane & 15;
  const int kb   = (lane < 16) ? 0 : 16;   // K half held by this lane (B layout)

  // --- B fragments in registers (one-time, tiny & L2 resident) -------------
  v16h Bd[4], Bf[4];
  #pragma unroll
  for (int n = 0; n < 4; ++n) {
    const int col = n*16 + lh;
    #pragma unroll
    for (int i = 0; i < 16; ++i) {
      Bd[n][i] = Wd1h  [(kb + i)*64 + col];
      Bf[n][i] = Wfoldh[(kb + i)*64 + col];
    }
  }

  // --- ray setup (uniform per wave) ----------------------------------------
  const float ox = rays[ray*6+0], oy = rays[ray*6+1], oz = rays[ray*6+2];
  const float dx = rays[ray*6+3], dy = rays[ray*6+4], dz = rays[ray*6+5];

  auto axis_t = [](float o, float d) -> float {
    const float v = (d == 0.0f) ? 1e-6f : d;
    const float a = ( 1.5f - o) / v;
    const float b = (-1.5f - o) / v;
    return fminf(a, b);
  };
  float tmin = fmaxf(fmaxf(axis_t(ox, dx), axis_t(oy, dy)), axis_t(oz, dz));
  tmin = fminf(fmaxf(tmin, 0.05f), 6.0f);

  // --- accumulators ---------------------------------------------------------
  float accR = 0.f, accG = 0.f, accB = 0.f, accD = 0.f, accO = 0.f;
  float lossU = 0.f, lossB = 0.f;
  float accSem[20]; float accI[16];
  #pragma unroll
  for (int j = 0; j < 20; ++j) accSem[j] = 0.f;
  #pragma unroll
  for (int j = 0; j < 16; ++j) accI[j] = 0.f;

  float carryT = 1.0f, carryW = 0.0f, carryWM = 0.0f;

  for (int c = 0; c < NCHUNK; ++c) {
    __syncthreads();   // uniform trip count across all waves -> safe

    const int   s  = c*CHUNK + lane;
    const float z  = tmin + (float)s * STEPF;
    const float px = ox + dx*z, py = oy + dy*z, pz = oz + dz*z;
    const bool  inb = (s < NSAMP) &&
                      px >= -1.5f && px <= 1.5f &&
                      py >= -1.5f && py <= 1.5f &&
                      pz >= -1.5f && pz <= 1.5f;
    const float xn = (px + 1.5f) * (2.0f/3.0f) - 1.0f;
    const float yn = (py + 1.5f) * (2.0f/3.0f) - 1.0f;
    const float zn = (pz + 1.5f) * (2.0f/3.0f) - 1.0f;

    // --- WMMA: both 64-wide layers for the two 16-row tiles of this chunk --
    #pragma unroll
    for (int t = 0; t < 2; ++t) {
      const int src = (t << 4) | lh;
      const float sx = __shfl(xn, src);
      const float sy = __shfl(yn, src);
      const float sz = __shfl(zn, src);

      // 16x32 f16 A: only K rows 0..5 nonzero -> lanes 0..15, elems 0..5
      v16h A = {};
      if (lane < 16) {
        A[0] = (_Float16)sx; A[1] = (_Float16)sy; A[2] = (_Float16)sz;
        A[3] = (_Float16)dx; A[4] = (_Float16)dy; A[5] = (_Float16)dz;
      }

      #pragma unroll
      for (int n = 0; n < 4; ++n) {
        v8f cd = {};
        cd = __builtin_amdgcn_wmma_f32_16x16x32_f16(false, A, false, Bd[n],
                                                    (short)0, cd, false, false);
        v8f cf = {};
        cf = __builtin_amdgcn_wmma_f32_16x16x32_f16(false, A, false, Bf[n],
                                                    (short)0, cf, false, false);
        const int   col = n*16 + lh;
        const int   rb  = t*16 + ((lane < 16) ? 0 : 8);
        const float b1  = bd1[col];
        const float b2  = ba1[col];
        #pragma unroll
        for (int v = 0; v < 8; ++v) {
          Hden[wv][rb + v][col] = (_Float16)fmaxf(cd[v] + b1, 0.0f);
          Happ[wv][rb + v][col] = (_Float16)fmaxf(cf[v] + b2, 0.0f);
        }
      }
    }
    __syncthreads();

    // --- density head: sigma = softplus(h . Wd2 + bd2), masked --------------
    float sp = bd2[0];
    #pragma unroll 8
    for (int k = 0; k < 64; ++k) sp += (float)Hden[wv][lane][k] * Wd2[k];
    float sigma = 0.0f;
    if (inb) sigma = (sp > 20.0f) ? sp : log1pf(expf(sp));

    const float dist  = (s >= NSAMP - 1) ? 0.0f : STEPF;
    const float alpha = 1.0f - expf(-sigma * dist * DSCALE);

    // --- transmittance: wave-wide inclusive product scan + carry ------------
    const float p = 1.0f - alpha + 1e-10f;
    float P = p;
    #pragma unroll
    for (int o = 1; o < 32; o <<= 1) {
      const float t2 = __shfl_up(P, o, 32);
      if (lane >= o) P *= t2;
    }
    float ex = __shfl_up(P, 1, 32);
    const float T = carryT * ((lane == 0) ? 1.0f : ex);
    const float w = alpha * T;
    carryT *= __shfl(P, 31, 32);

    // --- distortion loss: exclusive cumsums via shuffle scans ---------------
    const float midz = z + 0.5f * STEPF;
    const float wm   = w * midz;
    float Sw = w, Swm = wm;
    #pragma unroll
    for (int o = 1; o < 32; o <<= 1) {
      const float a2 = __shfl_up(Sw,  o, 32);
      const float b2 = __shfl_up(Swm, o, 32);
      if (lane >= o) { Sw += a2; Swm += b2; }
    }
    const float exw  = carryW  + (Sw  - w);
    const float exwm = carryWM + (Swm - wm);
    lossB += wm * exw - w * exwm;
    lossU += dist * w * w;
    carryW  += __shfl(Sw,  31, 32);
    carryWM += __shfl(Swm, 31, 32);

    accD += w * z;
    accO += w;

    // --- appearance / semantics / instances (gated by app_mask) -------------
    if (w > WTHRES) {
      float r0 = ba2[0], r1 = ba2[1], r2 = ba2[2];
      #pragma unroll 8
      for (int k = 0; k < 64; ++k) {
        const float h = (float)Happ[wv][lane][k];
        r0 += h * Wa2[k*3 + 0];
        r1 += h * Wa2[k*3 + 1];
        r2 += h * Wa2[k*3 + 2];
      }
      accR += w / (1.0f + expf(-r0));
      accG += w / (1.0f + expf(-r1));
      accB += w / (1.0f + expf(-r2));
      #pragma unroll
      for (int j = 0; j < 20; ++j)
        accSem[j] += w * (xn*Wcomb[j] + yn*Wcomb[20 + j] + zn*Wcomb[40 + j] + bsm[j]);
      #pragma unroll
      for (int j = 0; j < 16; ++j)
        accI[j] += w * (xn*Wi[j] + yn*Wi[16 + j] + zn*Wi[32 + j]);
    }
  }

  // --- wave reduction and output -------------------------------------------
  accR = wred(accR); accG = wred(accG); accB = wred(accB);
  accD = wred(accD); accO = wred(accO);
  lossU = wred(lossU); lossB = wred(lossB);
  #pragma unroll
  for (int j = 0; j < 20; ++j) accSem[j] = wred(accSem[j]);
  #pragma unroll
  for (int j = 0; j < 16; ++j) accI[j] = wred(accI[j]);

  if (lane == 0) {
    const float wb = (*white_bg != 0) ? 1.0f : 0.0f;
    out[ray*3 + 0] = fminf(fmaxf(accR + wb*(1.0f - accO), 0.0f), 1.0f);
    out[ray*3 + 1] = fminf(fmaxf(accG + wb*(1.0f - accO), 0.0f), 1.0f);
    out[ray*3 + 2] = fminf(fmaxf(accB + wb*(1.0f - accO), 0.0f), 1.0f);
    #pragma unroll
    for (int j = 0; j < 20; ++j) out[SEM_BASE  + ray*20 + j] = accSem[j];
    #pragma unroll
    for (int j = 0; j < 16; ++j) out[INST_BASE + ray*16 + j] = accI[j];
    out[DEPTH_BASE + ray] = accD;
    atomicAdd(out + DREG_IDX,
              (lossU * (1.0f/3.0f) + 2.0f * lossB) * (1.0f / (float)NRAYS));
  }
}

// ---------------------------------------------------------------------------
extern "C" void kernel_launch(void* const* d_in, const int* in_sizes, int n_in,
                              void* d_out, int out_size, void* d_ws, size_t ws_size,
                              hipStream_t stream) {
  (void)in_sizes; (void)n_in; (void)out_size; (void)ws_size;
  const float* rays = (const float*)d_in[0];
  const float* Wd1  = (const float*)d_in[1];
  const float* bd1  = (const float*)d_in[2];
  const float* Wd2  = (const float*)d_in[3];
  const float* bd2  = (const float*)d_in[4];
  const float* Wapp = (const float*)d_in[5];
  const float* Wa1  = (const float*)d_in[6];
  const float* ba1  = (const float*)d_in[7];
  const float* Wa2  = (const float*)d_in[8];
  const float* ba2  = (const float*)d_in[9];
  const float* Wsf  = (const float*)d_in[10];
  const float* Wsm  = (const float*)d_in[11];
  const float* bsm  = (const float*)d_in[12];
  const float* Wi   = (const float*)d_in[13];
  const int*   wbg  = (const int*)  d_in[15];
  float* out = (float*)d_out;

  _Float16* Wd1h   = (_Float16*)d_ws;                         // 2048 f16 = 4 KB
  _Float16* Wfoldh = (_Float16*)((char*)d_ws + 4096);         // 2048 f16 = 4 KB
  float*    Wcomb  = (float*)   ((char*)d_ws + 8192);         // 60 f32

  prep_kernel<<<1, 256, 0, stream>>>(Wd1, Wapp, Wa1, Wsf, Wsm,
                                     Wd1h, Wfoldh, Wcomb, out + DREG_IDX);
  render_kernel<<<NRAYS / WAVES, WAVES * 32, 0, stream>>>(
      rays, bd1, Wd2, bd2, ba1, Wa2, ba2, bsm, Wi,
      Wd1h, Wfoldh, Wcomb, wbg, out);
}
